// KWClipAlignmentBranch_51307679318743
// MI455X (gfx1250) — matrix operand order repaired and checked by reading, not compiled, not run on hardware
//
#include <hip/hip_runtime.h>

// ---------------- problem constants (match reference) ----------------
#define B_   16
#define T_   1024
#define D_   768
#define S_   64
#define H_   8
#define DH_  96
#define FF_  3072
#define TD_  512
#define V_   49408
#define ROWS_ (B_*S_)   // 1024 keyword rows

typedef __attribute__((ext_vector_type(16))) _Float16 v16h;
typedef __attribute__((ext_vector_type(8)))  float    v8f;

union V16H { v16h v; uint4 q[2]; _Float16 h[16]; };
union V8F  { v8f  v; float f[8]; };

__device__ __forceinline__ v8f wmma_f16(v16h a, v16h b, v8f c) {
  // D = A(16x32 f16) * B(32x16 f16) + C(16x16 f32)
  return __builtin_amdgcn_wmma_f32_16x16x32_f16(false, a, false, b, (short)0, c,
                                                false, false);
}

// A fragment (16x32, f16): lane r=lane&15 is row M=r; hh=lane>>4 selects K-halves.
// ISA layout: VGPR0-3 hold K = hh*8 + 0..7, VGPR4-7 hold K = 16 + hh*8 + 0..7.
__device__ __forceinline__ v16h frag_a_f16(const _Float16* rowp, int k, int hh) {
  V16H u;
  u.q[0] = *(const uint4*)(rowp + k + hh * 8);
  u.q[1] = *(const uint4*)(rowp + k + 16 + hh * 8);
  return u.v;
}
// B fragment (32x16, f16): lane r holds column N=r; elements 0-7 hold K = hh*8 + 0..7,
// elements 8-15 hold K = 16 + hh*8 + 0..7 (same K order as the A fragment).
__device__ __forceinline__ v16h frag_b_f16(const _Float16* rowp, int k, int hh) {
  V16H u;
  u.q[0] = *(const uint4*)(rowp + k + hh * 8);
  u.q[1] = *(const uint4*)(rowp + k + 16 + hh * 8);
  return u.v;
}
// A fragment built from f32 source (softmax probabilities in LDS), cvt to f16.
__device__ __forceinline__ v16h frag_a_f32(const float* rowp, int k, int hh) {
  V16H u;
  const float* p0 = rowp + k + hh * 8;
  const float* p1 = rowp + k + 16 + hh * 8;
#pragma unroll
  for (int j = 0; j < 8; ++j) {
    u.h[j]     = (_Float16)p0[j];
    u.h[j + 8] = (_Float16)p1[j];
  }
  return u.v;
}

// ---------------- weight transpose + f16 convert: w[K][N] -> wT[N][K] ----------
__global__ void prep_wT_kernel(const float* __restrict__ w, _Float16* __restrict__ wT,
                               int K, int N) {
  int idx = blockIdx.x * 256 + threadIdx.x;
  if (idx >= N * K) return;
  int n = idx / K, k = idx - n * K;
  wT[idx] = (_Float16)w[(size_t)k * N + n];
}

// ---------------- embedding prep: normalized rows (f16) + raw transpose (f16) --
__global__ __launch_bounds__(256) void prep_emb_kernel(const float* __restrict__ emb,
                                                       _Float16* __restrict__ embn,
                                                       _Float16* __restrict__ embT) {
  int v = blockIdx.x, tid = threadIdx.x;
  size_t base = (size_t)v * TD_;
  float a = emb[base + tid], b = emb[base + tid + 256];
  __shared__ float red[256];
  __shared__ float sh_inv;
  red[tid] = a * a + b * b;
  __syncthreads();
  for (int o = 128; o > 0; o >>= 1) { if (tid < o) red[tid] += red[tid + o]; __syncthreads(); }
  if (tid == 0) sh_inv = 1.0f / fmaxf(sqrtf(red[0]), 1e-8f);
  __syncthreads();
  float inv = sh_inv;
  embn[base + tid]       = (_Float16)(a * inv);
  embn[base + tid + 256] = (_Float16)(b * inv);
  embT[(size_t)tid * V_ + v]         = (_Float16)a;
  embT[(size_t)(tid + 256) * V_ + v] = (_Float16)b;
}

// ---------------- segment mean pool + LayerNorm1 ------------------------------
__global__ __launch_bounds__(256) void pool_ln_kernel(
    const float* __restrict__ audio, const int* __restrict__ audio_len,
    const int* __restrict__ aligns, const int* __restrict__ anum,
    const float* __restrict__ g, const float* __restrict__ bb,
    float* __restrict__ o32, _Float16* __restrict__ o16) {
  int bs = blockIdx.x;            // b*64 + s
  int b = bs >> 6, s = bs & 63;
  int tid = threadIdx.x;
  int ts = aligns[bs * 2 + 0], te = aligns[bs * 2 + 1];
  int an = anum[b], al = audio_len[b];
  bool valid = (ts >= 0) && (s < an);
  int te_eff = (te == al) ? T_ : te;
  int tsc = min(max(ts, 0), T_);
  int tec = min(max(te_eff, 0), T_);
  float cnt = (float)max(tec - tsc, 1);
  float x0 = 0.f, x1 = 0.f, x2 = 0.f;
  if (valid) {
    for (int t = tsc; t < tec; ++t) {
      const float* row = audio + ((size_t)b * T_ + t) * D_;
      x0 += row[tid]; x1 += row[tid + 256]; x2 += row[tid + 512];
    }
    x0 /= cnt; x1 /= cnt; x2 /= cnt;
  }
  __shared__ float red[256];
  __shared__ float sh_m, sh_r;
  red[tid] = x0 + x1 + x2;
  __syncthreads();
  for (int o = 128; o > 0; o >>= 1) { if (tid < o) red[tid] += red[tid + o]; __syncthreads(); }
  if (tid == 0) sh_m = red[0] / (float)D_;
  __syncthreads();
  float m = sh_m;
  float d0 = x0 - m, d1 = x1 - m, d2 = x2 - m;
  red[tid] = d0 * d0 + d1 * d1 + d2 * d2;
  __syncthreads();
  for (int o = 128; o > 0; o >>= 1) { if (tid < o) red[tid] += red[tid + o]; __syncthreads(); }
  if (tid == 0) sh_r = rsqrtf(red[0] / (float)D_ + 1e-5f);
  __syncthreads();
  float rs = sh_r;
  size_t base = (size_t)bs * D_;
  float y0 = d0 * rs * g[tid] + bb[tid];
  float y1 = d1 * rs * g[tid + 256] + bb[tid + 256];
  float y2 = d2 * rs * g[tid + 512] + bb[tid + 512];
  o32[base + tid] = y0; o32[base + tid + 256] = y1; o32[base + tid + 512] = y2;
  o16[base + tid] = (_Float16)y0; o16[base + tid + 256] = (_Float16)y1;
  o16[base + tid + 512] = (_Float16)y2;
}

// ---------------- residual + LayerNorm (D_=768) -------------------------------
__global__ __launch_bounds__(256) void ln_res_kernel(
    const float* __restrict__ a, const float* __restrict__ res,
    const float* __restrict__ g, const float* __restrict__ bb,
    float* __restrict__ o32, _Float16* __restrict__ o16) {
  int row = blockIdx.x, tid = threadIdx.x;
  size_t base = (size_t)row * D_;
  float x0 = a[base + tid]       + (res ? res[base + tid] : 0.f);
  float x1 = a[base + tid + 256] + (res ? res[base + tid + 256] : 0.f);
  float x2 = a[base + tid + 512] + (res ? res[base + tid + 512] : 0.f);
  __shared__ float red[256];
  __shared__ float sh_m, sh_r;
  red[tid] = x0 + x1 + x2;
  __syncthreads();
  for (int o = 128; o > 0; o >>= 1) { if (tid < o) red[tid] += red[tid + o]; __syncthreads(); }
  if (tid == 0) sh_m = red[0] / (float)D_;
  __syncthreads();
  float m = sh_m;
  float d0 = x0 - m, d1 = x1 - m, d2 = x2 - m;
  red[tid] = d0 * d0 + d1 * d1 + d2 * d2;
  __syncthreads();
  for (int o = 128; o > 0; o >>= 1) { if (tid < o) red[tid] += red[tid + o]; __syncthreads(); }
  if (tid == 0) sh_r = rsqrtf(red[0] / (float)D_ + 1e-5f);
  __syncthreads();
  float rs = sh_r;
  float y0 = d0 * rs * g[tid] + bb[tid];
  float y1 = d1 * rs * g[tid + 256] + bb[tid + 256];
  float y2 = d2 * rs * g[tid + 512] + bb[tid + 512];
  if (o32) { o32[base + tid] = y0; o32[base + tid + 256] = y1; o32[base + tid + 512] = y2; }
  if (o16) { o16[base + tid] = (_Float16)y0; o16[base + tid + 256] = (_Float16)y1;
             o16[base + tid + 512] = (_Float16)y2; }
}

// ---------------- generic WMMA GEMM: C[M,N] = A[M,K] @ BT[N,K]^T + bias --------
// One wave computes a 32x64 macro-tile (2 M-tiles x 4 N-tiles): per K-step,
// 6 fragment loads feed 8 v_wmma ops (2.7x better load:mma ratio than 1x1).
__global__ __launch_bounds__(128) void gemm_kernel(
    const _Float16* __restrict__ A, const _Float16* __restrict__ BT,
    const float* __restrict__ bias,
    float* __restrict__ C32, _Float16* __restrict__ C16,
    int M, int N, int K, int relu) {
  int wave = blockIdx.x * 4 + (threadIdx.x >> 5);
  int lane = threadIdx.x & 31;
  int tilesN = N >> 6;                  // 64-wide N groups
  int tm = wave / tilesN;               // 32-row group
  int tn = wave - tm * tilesN;          // 64-col group
  int r = lane & 15, hh = lane >> 4;
  const _Float16* arow0 = A + (size_t)(tm * 32 + r) * K;
  const _Float16* arow1 = arow0 + (size_t)16 * K;
  const _Float16* brow0 = BT + (size_t)(tn * 64 + r) * K;
  const _Float16* brow1 = brow0 + (size_t)16 * K;
  const _Float16* brow2 = brow0 + (size_t)32 * K;
  const _Float16* brow3 = brow0 + (size_t)48 * K;
  V8F c[2][4];
#pragma unroll
  for (int mi = 0; mi < 2; ++mi)
#pragma unroll
    for (int t = 0; t < 4; ++t)
#pragma unroll
      for (int i = 0; i < 8; ++i) c[mi][t].f[i] = 0.f;

  for (int k = 0; k < K; k += 32) {
    v16h a0 = frag_a_f16(arow0, k, hh);
    v16h a1 = frag_a_f16(arow1, k, hh);
    v16h b0 = frag_b_f16(brow0, k, hh);
    v16h b1 = frag_b_f16(brow1, k, hh);
    v16h b2 = frag_b_f16(brow2, k, hh);
    v16h b3 = frag_b_f16(brow3, k, hh);
    c[0][0].v = wmma_f16(a0, b0, c[0][0].v);
    c[0][1].v = wmma_f16(a0, b1, c[0][1].v);
    c[0][2].v = wmma_f16(a0, b2, c[0][2].v);
    c[0][3].v = wmma_f16(a0, b3, c[0][3].v);
    c[1][0].v = wmma_f16(a1, b0, c[1][0].v);
    c[1][1].v = wmma_f16(a1, b1, c[1][1].v);
    c[1][2].v = wmma_f16(a1, b2, c[1][2].v);
    c[1][3].v = wmma_f16(a1, b3, c[1][3].v);
  }
#pragma unroll
  for (int t = 0; t < 4; ++t) {
    int gn = tn * 64 + t * 16 + r;
    float bv = bias ? bias[gn] : 0.f;
#pragma unroll
    for (int mi = 0; mi < 2; ++mi) {
#pragma unroll
      for (int i = 0; i < 8; ++i) {
        int gm = tm * 32 + mi * 16 + hh * 8 + i;  // D layout: VGPR i -> row i (+8 hi half)
        float val = c[mi][t].f[i] + bv;
        if (relu) val = fmaxf(val, 0.f);
        size_t off = (size_t)gm * N + gn;
        if (C32) C32[off] = val;
        if (C16) C16[off] = (_Float16)val;
      }
    }
  }
}

// ---------------- masked MHA over S=64 segments (tiny; VALU) -------------------
__global__ __launch_bounds__(64) void attn_kernel(
    const float* __restrict__ q, const float* __restrict__ k,
    const float* __restrict__ v, const int* __restrict__ anum,
    _Float16* __restrict__ ctx16) {
  int bh = blockIdx.x;
  int b = bh >> 3, h = bh & 7;
  int qi = threadIdx.x;
  int an = anum[b];
  const float scale = 0.1020620726159658f;   // 1/sqrt(96)
  float sc[S_];
  const float* qrow = q + ((size_t)(b * S_ + qi)) * D_ + h * DH_;
  float mx = -1e30f;
  for (int kk = 0; kk < S_; ++kk) {
    const float* krow = k + ((size_t)(b * S_ + kk)) * D_ + h * DH_;
    float d = 0.f;
    for (int dd = 0; dd < DH_; ++dd) d += qrow[dd] * krow[dd];
    d *= scale;
    if (kk >= an) d = -1e30f;
    sc[kk] = d;
    mx = fmaxf(mx, d);
  }
  float sum = 0.f;
  for (int kk = 0; kk < S_; ++kk) { float e = __expf(sc[kk] - mx); sc[kk] = e; sum += e; }
  float inv = 1.f / sum;
  _Float16* orow = ctx16 + ((size_t)(b * S_ + qi)) * D_ + h * DH_;
  for (int dd = 0; dd < DH_; ++dd) {
    float a = 0.f;
    for (int kk = 0; kk < S_; ++kk)
      a += sc[kk] * v[((size_t)(b * S_ + kk)) * D_ + h * DH_ + dd];
    orow[dd] = (_Float16)(a * inv);
  }
}

// ---------------- kw row-normalize -> f16 --------------------------------------
__global__ __launch_bounds__(256) void rownorm_kernel(const float* __restrict__ kw,
                                                      _Float16* __restrict__ out) {
  int row = blockIdx.x, tid = threadIdx.x;
  size_t base = (size_t)row * TD_;
  float a = kw[base + tid], b = kw[base + tid + 256];
  __shared__ float red[256];
  __shared__ float sh_inv;
  red[tid] = a * a + b * b;
  __syncthreads();
  for (int o = 128; o > 0; o >>= 1) { if (tid < o) red[tid] += red[tid + o]; __syncthreads(); }
  if (tid == 0) sh_inv = 1.0f / fmaxf(sqrtf(red[0]), 1e-8f);
  __syncthreads();
  float inv = sh_inv;
  out[base + tid]       = (_Float16)(a * inv);
  out[base + tid + 256] = (_Float16)(b * inv);
}

// ---------------- fused vocab stage (flash-softmax + two WMMA GEMMs) -----------
// Block = 256 threads (8 waves), handles 16 keyword rows. Streams vocab in
// chunks of 256: scores (WMMA, K=512) -> online softmax -> acc += p @ emb
// (WMMA, K=256) with per-wave 16x64 f32 register accumulators. No 200MB
// score buffer ever touches HBM; token_emb (f16, ~100MB both layouts) stays
// L2-resident (192MB) and the next chunk is prefetched (global_prefetch_b8)
// while the current chunk's 64 WMMAs execute.
__global__ __launch_bounds__(256) void vocab_kernel(
    const _Float16* __restrict__ kwf16,   // [1024][512] normalized rows
    const _Float16* __restrict__ embn,    // [V][512]    normalized, f16
    const _Float16* __restrict__ embT,    // [512][V]    raw, f16, transposed
    float* __restrict__ out) {            // [1024][512]
  __shared__ _Float16 kws[16 * TD_];      // 16 KB
  __shared__ float sbuf[16 * 256];        // 16 KB: scores -> probabilities
  __shared__ float mrow[16], lrow[16], srow[16];
  __shared__ float red[16][17];
  int tid = threadIdx.x;
  int lane = tid & 31, wv = tid >> 5;
  int r = lane & 15, hh = lane >> 4;
  int rowbase = blockIdx.x * 16;

  {   // stage the 16 kw rows into LDS (16KB, 1024 uint4)
    const uint4* src = (const uint4*)(kwf16 + (size_t)rowbase * TD_);
    uint4* dst = (uint4*)kws;
    for (int i = tid; i < (16 * TD_) / 8; i += 256) dst[i] = src[i];
  }
  if (tid < 16) { mrow[tid] = -1e30f; lrow[tid] = 0.f; }
  __syncthreads();

  V8F acc[4];
#pragma unroll
  for (int t = 0; t < 4; ++t)
#pragma unroll
    for (int i = 0; i < 8; ++i) acc[t].f[i] = 0.f;

  const int rr = tid >> 4;   // softmax bookkeeping: 16 threads per row
  const int ii = tid & 15;

  for (int cbase = 0; cbase < V_; cbase += 256) {
    // ---- phase 1: scores for this chunk (each wave: 16 x 32 cols, K=512)
    {
      int c0 = cbase + wv * 32;
      const _Float16* ar  = kws + r * TD_;
      const _Float16* b0p = embn + (size_t)(c0 + r) * TD_;
      const _Float16* b1p = embn + (size_t)(c0 + 16 + r) * TD_;
      if (cbase + 256 < V_) {           // uniform branch: L2 prefetch next chunk
        __builtin_prefetch(b0p + (size_t)256 * TD_, 0, 2);
        __builtin_prefetch(b1p + (size_t)256 * TD_, 0, 2);
        __builtin_prefetch(b0p + (size_t)256 * TD_ + 256, 0, 2);
        __builtin_prefetch(b1p + (size_t)256 * TD_ + 256, 0, 2);
      }
      V8F s0, s1;
#pragma unroll
      for (int i = 0; i < 8; ++i) { s0.f[i] = 0.f; s1.f[i] = 0.f; }
      for (int k = 0; k < TD_; k += 32) {
        v16h a = frag_a_f16(ar, k, hh);
        s0.v = wmma_f16(a, frag_b_f16(b0p, k, hh), s0.v);
        s1.v = wmma_f16(a, frag_b_f16(b1p, k, hh), s1.v);
      }
#pragma unroll
      for (int i = 0; i < 8; ++i) {
        int row = hh * 8 + i;
        sbuf[row * 256 + wv * 32 + r]      = s0.f[i];
        sbuf[row * 256 + wv * 32 + 16 + r] = s1.f[i];
      }
    }
    __syncthreads();
    // ---- phase 2: online softmax bookkeeping (block-wide, in LDS)
    {
      float lm = -1e30f;
      for (int j = ii; j < 256; j += 16) lm = fmaxf(lm, sbuf[rr * 256 + j]);
      red[rr][ii] = lm;
      __syncthreads();
      if (ii == 0) {
        float cm = red[rr][0];
        for (int j = 1; j < 16; ++j) cm = fmaxf(cm, red[rr][j]);
        float mn = fmaxf(mrow[rr], cm);
        srow[rr] = __expf(mrow[rr] - mn);
        mrow[rr] = mn;
      }
      __syncthreads();
      float mr = mrow[rr];
      float ps = 0.f;
      for (int j = ii; j < 256; j += 16) {
        float p = __expf(sbuf[rr * 256 + j] - mr);
        sbuf[rr * 256 + j] = p;
        ps += p;
      }
      red[rr][ii] = ps;
      __syncthreads();
      if (ii == 0) {
        float s = 0.f;
        for (int j = 0; j < 16; ++j) s += red[rr][j];
        lrow[rr] = lrow[rr] * srow[rr] + s;
      }
      __syncthreads();
    }
    // ---- phase 3: acc = acc*scale + p[16x256] @ embT_chunk[256x512]
    {
#pragma unroll
      for (int t = 0; t < 4; ++t)
#pragma unroll
        for (int i = 0; i < 8; ++i) acc[t].f[i] *= srow[hh * 8 + i];
      const _Float16* bp0 = embT + (size_t)(wv * 64 +  0 + r) * V_ + cbase;
      const _Float16* bp1 = embT + (size_t)(wv * 64 + 16 + r) * V_ + cbase;
      const _Float16* bp2 = embT + (size_t)(wv * 64 + 32 + r) * V_ + cbase;
      const _Float16* bp3 = embT + (size_t)(wv * 64 + 48 + r) * V_ + cbase;
      if (cbase + 256 < V_) {           // uniform branch: L2 prefetch next chunk
        __builtin_prefetch(bp0 + 256, 0, 2);
        __builtin_prefetch(bp1 + 256, 0, 2);
        __builtin_prefetch(bp2 + 256, 0, 2);
        __builtin_prefetch(bp3 + 256, 0, 2);
      }
      const float* prow = &sbuf[r * 256];
      for (int kv = 0; kv < 256; kv += 32) {
        v16h a = frag_a_f32(prow, kv, hh);
        acc[0].v = wmma_f16(a, frag_b_f16(bp0, kv, hh), acc[0].v);
        acc[1].v = wmma_f16(a, frag_b_f16(bp1, kv, hh), acc[1].v);
        acc[2].v = wmma_f16(a, frag_b_f16(bp2, kv, hh), acc[2].v);
        acc[3].v = wmma_f16(a, frag_b_f16(bp3, kv, hh), acc[3].v);
      }
    }
    __syncthreads();
  }
  // ---- writeout: keywords = acc / l
#pragma unroll
  for (int t = 0; t < 4; ++t) {
    int gc = wv * 64 + t * 16 + r;
#pragma unroll
    for (int i = 0; i < 8; ++i) {
      int row = hh * 8 + i;
      float l = lrow[row];
      out[(size_t)(rowbase + row) * TD_ + gc] = (l > 0.f) ? acc[t].f[i] / l : 0.f;
    }
  }
}

// ---------------- host-side orchestration --------------------------------------
extern "C" void kernel_launch(void* const* d_in, const int* in_sizes, int n_in,
                              void* d_out, int out_size, void* d_ws, size_t ws_size,
                              hipStream_t stream) {
  const float* audio  = (const float*)d_in[0];
  const float* ln1_g  = (const float*)d_in[1];
  const float* ln1_b  = (const float*)d_in[2];
  const float* wq     = (const float*)d_in[3];
  const float* bq     = (const float*)d_in[4];
  const float* wk     = (const float*)d_in[5];
  const float* bk     = (const float*)d_in[6];
  const float* wv     = (const float*)d_in[7];
  const float* bv     = (const float*)d_in[8];
  const float* wo     = (const float*)d_in[9];
  const float* bo     = (const float*)d_in[10];
  const float* lna_g  = (const float*)d_in[11];
  const float* lna_b  = (const float*)d_in[12];
  const float* w1     = (const float*)d_in[13];
  const float* b1     = (const float*)d_in[14];
  const float* w2     = (const float*)d_in[15];
  const float* b2     = (const float*)d_in[16];
  const float* lnb_g  = (const float*)d_in[17];
  const float* lnb_b  = (const float*)d_in[18];
  const float* wp     = (const float*)d_in[19];
  const float* bp     = (const float*)d_in[20];
  const float* emb    = (const float*)d_in[21];
  const int* audio_len = (const int*)d_in[22];
  const int* aligns    = (const int*)d_in[23];
  const int* anum      = (const int*)d_in[24];
  float* out = (float*)d_out;

  char* ws = (char*)d_ws;
  size_t off = 0;
  auto alloc = [&](size_t bytes) -> void* {
    void* p = ws + off;
    off += (bytes + 255) & ~(size_t)255;
    return p;
  };
  _Float16* wqT  = (_Float16*)alloc((size_t)D_ * D_ * 2);
  _Float16* wkT  = (_Float16*)alloc((size_t)D_ * D_ * 2);
  _Float16* wvT  = (_Float16*)alloc((size_t)D_ * D_ * 2);
  _Float16* woT  = (_Float16*)alloc((size_t)D_ * D_ * 2);
  _Float16* w1T  = (_Float16*)alloc((size_t)FF_ * D_ * 2);
  _Float16* w2T  = (_Float16*)alloc((size_t)D_ * FF_ * 2);
  _Float16* wpT  = (_Float16*)alloc((size_t)TD_ * D_ * 2);
  _Float16* embn16 = (_Float16*)alloc((size_t)V_ * TD_ * 2);
  _Float16* embT16 = (_Float16*)alloc((size_t)TD_ * V_ * 2);
  float*    src32  = (float*)alloc((size_t)ROWS_ * D_ * 4);
  _Float16* src16  = (_Float16*)alloc((size_t)ROWS_ * D_ * 2);
  float*    q32    = (float*)alloc((size_t)ROWS_ * D_ * 4);
  float*    k32    = (float*)alloc((size_t)ROWS_ * D_ * 4);
  float*    v32    = (float*)alloc((size_t)ROWS_ * D_ * 4);
  _Float16* ctx16  = (_Float16*)alloc((size_t)ROWS_ * D_ * 2);
  float*    atn32  = (float*)alloc((size_t)ROWS_ * D_ * 4);
  float*    x1_32  = (float*)alloc((size_t)ROWS_ * D_ * 4);
  _Float16* x1_16  = (_Float16*)alloc((size_t)ROWS_ * D_ * 2);
  _Float16* hff16  = (_Float16*)alloc((size_t)ROWS_ * FF_ * 2);
  float*    ffo32  = (float*)alloc((size_t)ROWS_ * D_ * 4);
  _Float16* x2_16  = (_Float16*)alloc((size_t)ROWS_ * D_ * 2);
  float*    kw32   = (float*)alloc((size_t)ROWS_ * TD_ * 4);
  _Float16* kwf16  = (_Float16*)alloc((size_t)ROWS_ * TD_ * 2);

  auto launch_wT = [&](const float* w_, _Float16* wT_, int K, int N) {
    int total = K * N;
    prep_wT_kernel<<<(total + 255) / 256, 256, 0, stream>>>(w_, wT_, K, N);
  };
  launch_wT(wq, wqT, D_, D_);
  launch_wT(wk, wkT, D_, D_);
  launch_wT(wv, wvT, D_, D_);
  launch_wT(wo, woT, D_, D_);
  launch_wT(w1, w1T, D_, FF_);
  launch_wT(w2, w2T, FF_, D_);
  launch_wT(wp, wpT, D_, TD_);
  prep_emb_kernel<<<V_, 256, 0, stream>>>(emb, embn16, embT16);

  pool_ln_kernel<<<ROWS_, 256, 0, stream>>>(audio, audio_len, aligns, anum,
                                            ln1_g, ln1_b, src32, src16);

  // one wave per 32x64 macro-tile; 128-thread blocks (4 waves)
  auto gemm = [&](const _Float16* A, const _Float16* BT, const float* bias,
                  float* C32, _Float16* C16, int M, int N, int K, int relu) {
    int waves = (M / 32) * (N / 64);
    gemm_kernel<<<waves / 4, 128, 0, stream>>>(A, BT, bias, C32, C16, M, N, K, relu);
  };
  gemm(src16, wqT, bq, q32, nullptr, ROWS_, D_, D_, 0);
  gemm(src16, wkT, bk, k32, nullptr, ROWS_, D_, D_, 0);
  gemm(src16, wvT, bv, v32, nullptr, ROWS_, D_, D_, 0);

  attn_kernel<<<B_ * H_, 64, 0, stream>>>(q32, k32, v32, anum, ctx16);

  gemm(ctx16, woT, bo, atn32, nullptr, ROWS_, D_, D_, 0);
  ln_res_kernel<<<ROWS_, 256, 0, stream>>>(src32, atn32, lna_g, lna_b, x1_32, x1_16);

  gemm(x1_16, w1T, b1, nullptr, hff16, ROWS_, FF_, D_, 1);   // relu
  gemm(hff16, w2T, b2, ffo32, nullptr, ROWS_, D_, FF_, 0);
  ln_res_kernel<<<ROWS_, 256, 0, stream>>>(x1_32, ffo32, lnb_g, lnb_b, nullptr, x2_16);

  gemm(x2_16, wpT, bp, kw32, nullptr, ROWS_, TD_, D_, 0);
  rownorm_kernel<<<ROWS_, 256, 0, stream>>>(kw32, kwf16);

  vocab_kernel<<<ROWS_ / 16, 256, 0, stream>>>(kwf16, embn16, embT16, out);
}
